// LLM_MAIN_18923625906735
// MI455X (gfx1250) — matrix operand order repaired and checked by reading, not compiled, not run on hardware
//
#include <hip/hip_runtime.h>
#include <hip/hip_bf16.h>
#include <math.h>
#include <stdint.h>

// Problem constants (fixed by the reference)
#define S_    1024
#define D_    2048
#define NH_   16
#define NKV_  8
#define HD_   128
#define FF_   6144
#define V_    32000
#define L_    2
#define KV_   1024          // kv_len = history(0) + S
#define QKVW_ 4096          // (NH + 2*NKV) * HD

typedef __attribute__((ext_vector_type(16))) _Float16 v16h;
typedef __attribute__((ext_vector_type(8)))  _Float16 v8h;
typedef __attribute__((ext_vector_type(8)))  float    v8f;
typedef __attribute__((ext_vector_type(4)))  float    v4f;

// ---------------------------------------------------------------------------
// CDNA5 async global->LDS copy (GLOBAL_LOAD_ASYNC_TO_LDS_B128, ASYNCcnt).
// LDS destination = low 32 bits of the generic pointer (ISA: LDS addr is the
// truncated flat address). Per ISA pseudocode the instruction's INST_OFFSET
// is added to BOTH the LDS destination and the global source, so one base
// (lds, global) pair covers a 64-byte row chunk via offset: immediates.
__device__ __forceinline__ void async_row64(_Float16* lds, const _Float16* g) {
  uint32_t l = (uint32_t)(uintptr_t)lds;
  uint64_t a = (uint64_t)(uintptr_t)g;
  asm volatile(
      "global_load_async_to_lds_b128 %0, %1, off\n\t"
      "global_load_async_to_lds_b128 %0, %1, off offset:16\n\t"
      "global_load_async_to_lds_b128 %0, %1, off offset:32\n\t"
      "global_load_async_to_lds_b128 %0, %1, off offset:48"
      :: "v"(l), "v"(a) : "memory");
}
__device__ __forceinline__ void wait_async0() {
  asm volatile("s_wait_asynccnt 0x0" ::: "memory");
}

// ---------------------------------------------------------------------------
// WMMA fragment loaders for V_WMMA_F32_16X16X32_F16 (ISA 7.12.2).
// A (16x32, row-major source): lane = h*16 + r holds row r; v16h elements
//   e[0..7]  = k = h*8 + 0..7        (VGPR0-3, 2 halfs each)
//   e[8..15] = k = 16 + h*8 + 0..7   (VGPR4-7)
__device__ __forceinline__ v16h frag_a(const _Float16* p) {
  v8h lo = *(const v8h*)p;          // k = h*8 ..
  v8h hi = *(const v8h*)(p + 16);   // k = 16 + h*8 ..
  v16h f;
#pragma unroll
  for (int i = 0; i < 8; ++i) { f[i] = lo[i]; f[i + 8] = hi[i]; }
  return f;
}
// B (32x16): lane = h*16 + n holds column n; k = h*8 + 0..7, 16 + h*8 + 0..7
// (dense restriction of the ISA 7.12.4 B-matrix layout). Since we feed B as
// rows of W (C = A * W^T), a "column of B" is a row of W: same row-major load.
__device__ __forceinline__ v16h frag_b(const _Float16* p) {
  v8h lo = *(const v8h*)p;
  v8h hi = *(const v8h*)(p + 16);
  v16h f;
#pragma unroll
  for (int i = 0; i < 8; ++i) { f[i] = lo[i]; f[i + 8] = hi[i]; }
  return f;
}

// ---------------------------------------------------------------------------
// Generic WMMA GEMM:  C[M,N] = A[M,K] (f16) * B[N,K]^T
//   BF32   : B is fp32 (weights), converted to f16 on the fly while staging
//   RES    : C = Res + A*B^T (fp32 residual stream)
//   F16OUT : write f16 instead of fp32
// grid = (N/128, M/128, batch); per-batch element strides sA/sB/sC; B batch
// index = z / bDiv (GQA head sharing). K must be a multiple of 64.
// Block = 256 threads = 8 waves; wave computes a 32x64 subtile (2x4 x 16x16).
// A tiles (and f16 B tiles) are staged with GLOBAL_LOAD_ASYNC_TO_LDS_B128.
template <bool BF32, bool RES, bool F16OUT>
__global__ void __launch_bounds__(256)
wgemm(const _Float16* __restrict__ A, const void* __restrict__ Bv,
      void* __restrict__ Cv, const float* __restrict__ Res,
      int K, int lda, int ldb, int ldc,
      long sA, long sB, long sC, int bDiv) {
  constexpr int TKP = 72;                       // 64 k-halfs + 8 pad
  __shared__ _Float16 As[128 * TKP];
  __shared__ _Float16 Bs[128 * TKP];

  const int tid  = threadIdx.x;
  const int lane = tid & 31;
  const int wave = tid >> 5;
  const int z    = blockIdx.z;

  A += (long)z * sA;
  const _Float16* Bh = (const _Float16*)Bv;
  const float*    Bf = (const float*)Bv;
  const long boff = (long)(z / bDiv) * sB;
  if (BF32) Bf += boff; else Bh += boff;
  float*    Cf = (float*)Cv + (F16OUT ? 0 : (long)z * sC);
  _Float16* Ch = (_Float16*)Cv + (F16OUT ? (long)z * sC : 0);
  if (RES) Res += (long)z * sC;

  const int tm = blockIdx.y * 128;
  const int tn = blockIdx.x * 128;
  const int wm = (wave & 3) * 32;               // 4 wave-rows of 32
  const int wn = (wave >> 2) * 64;              // 2 wave-cols of 64

  v8f acc[2][4];
  {
    v8f zero = {};
#pragma unroll
    for (int i = 0; i < 2; ++i)
#pragma unroll
      for (int j = 0; j < 4; ++j) acc[i][j] = zero;
  }

  const int lr = tid >> 1;                      // tile row this thread stages
  const int lc = (tid & 1) * 32;                // k-offset 0 or 32

  for (int k0 = 0; k0 < K; k0 += 64) {
    // ---- stage A tile (128 x 64 f16): async global->LDS, no VGPR staging ----
    {
      const _Float16* srcA = A + (long)(tm + lr) * lda + (k0 + lc);
      async_row64(&As[lr * TKP + lc], srcA);
      if (k0 + 64 < K) __builtin_prefetch(srcA + 64, 0, 1);
    }
    // ---- stage B tile (128 x 64) ----
    if (BF32) {
      // fp32 weights: must pass through VALU for the f32->f16 convert.
      // Load all 8 quads first so the loads batch under one wait.
      const float* srcB = Bf + (long)(tn + lr) * ldb + (k0 + lc);
      _Float16* dstB = &Bs[lr * TKP + lc];
      v4f f[8];
#pragma unroll
      for (int q = 0; q < 8; ++q) f[q] = *(const v4f*)(srcB + q * 4);
#pragma unroll
      for (int q = 0; q < 4; ++q) {
        v8h hq;
#pragma unroll
        for (int i = 0; i < 4; ++i) {
          hq[i]     = (_Float16)f[2 * q][i];
          hq[i + 4] = (_Float16)f[2 * q + 1][i];
        }
        *(v8h*)(dstB + q * 8) = hq;
      }
      if (k0 + 64 < K) __builtin_prefetch(srcB + 64, 0, 1);
    } else {
      const _Float16* srcB = Bh + (long)(tn + lr) * ldb + (k0 + lc);
      async_row64(&Bs[lr * TKP + lc], srcB);
      if (k0 + 64 < K) __builtin_prefetch(srcB + 64, 0, 1);
    }
    wait_async0();          // our async copies done (compiler adds dscnt wait)
    __syncthreads();

    const int r = lane & 15;
    const int h = lane >> 4;
#pragma unroll
    for (int kk = 0; kk < 2; ++kk) {            // two 32-deep WMMA sweeps
      const int kc = kk * 32;
      v16h af[2], bf[4];
#pragma unroll
      for (int mi = 0; mi < 2; ++mi)
        af[mi] = frag_a(&As[(wm + mi * 16 + r) * TKP + kc + h * 8]);
#pragma unroll
      for (int ni = 0; ni < 4; ++ni)
        bf[ni] = frag_b(&Bs[(wn + ni * 16 + r) * TKP + kc + h * 8]);
#pragma unroll
      for (int mi = 0; mi < 2; ++mi)
#pragma unroll
        for (int ni = 0; ni < 4; ++ni)
          acc[mi][ni] = __builtin_amdgcn_wmma_f32_16x16x32_f16(
              false, af[mi], false, bf[ni], (short)0, acc[mi][ni], false, false);
    }
    __syncthreads();
  }

  // ---- branch-free epilogue: lane col = lane&15, rows (lane>>4)*8 + r ----
  const int cn  = lane & 15;
  const int mrb = (lane >> 4) * 8;
#pragma unroll
  for (int mi = 0; mi < 2; ++mi)
#pragma unroll
    for (int r = 0; r < 8; ++r) {
      const long row =
          (long)(tm + wm + mi * 16 + mrb + r) * ldc + (tn + wn + cn);
#pragma unroll
      for (int ni = 0; ni < 4; ++ni) {
        float v = acc[mi][ni][r];
        if constexpr (RES) v += Res[row + ni * 16];
        if constexpr (F16OUT) Ch[row + ni * 16] = (_Float16)v;
        else                  Cf[row + ni * 16] = v;
      }
    }
}

// ---------------------------------------------------------------------------
// RMSNorm over D=2048; writes f16 and/or fp32 copies of the normalized row.
__global__ void __launch_bounds__(256)
rmsnorm_kernel(const float* __restrict__ x, _Float16* __restrict__ o16,
               float* __restrict__ o32) {
  const int s = blockIdx.x;
  const float* row = x + (size_t)s * D_;
  __shared__ float red[256];
  float v[8], ss = 0.f;
#pragma unroll
  for (int i = 0; i < 8; ++i) {
    v[i] = row[threadIdx.x + i * 256];
    ss += v[i] * v[i];
  }
  red[threadIdx.x] = ss;
  __syncthreads();
  for (int st = 128; st > 0; st >>= 1) {
    if (threadIdx.x < st) red[threadIdx.x] += red[threadIdx.x + st];
    __syncthreads();
  }
  const float scale = rsqrtf(red[0] * (1.0f / D_) + 1e-6f);
#pragma unroll
  for (int i = 0; i < 8; ++i) {
    const int d = threadIdx.x + i * 256;
    const float y = v[i] * scale;
    if (o16) o16[(size_t)s * D_ + d] = (_Float16)y;
    if (o32) o32[(size_t)s * D_ + d] = y;
  }
}

// ---------------------------------------------------------------------------
// Split qkv row, apply RoPE (tables already carry HD^-0.25), lay out:
//   q16 [NH, S, HD] f16 ; kf [NKV, S, HD] f32 ; vf [NKV, S, HD] f32
__global__ void __launch_bounds__(256)
rope_split_kernel(const float* __restrict__ qkv, const float* __restrict__ cosT,
                  const float* __restrict__ sinT, const int* __restrict__ histp,
                  _Float16* __restrict__ q16, float* __restrict__ kf,
                  float* __restrict__ vf) {
  const int s = blockIdx.x;
  const int pos = s + *histp;
  const float* row = qkv + (size_t)s * QKVW_;
  const float* cs = cosT + (size_t)pos * HD_;
  const float* sn = sinT + (size_t)pos * HD_;
  for (int idx = threadIdx.x; idx < QKVW_; idx += 256) {
    const int d = idx & (HD_ - 1);
    if (idx < NH_ * HD_) {                                    // Q
      const int qh = idx >> 7;
      const float x = row[idx];
      const float xr = (d < 64) ? -row[qh * HD_ + d + 64] : row[qh * HD_ + d - 64];
      q16[((size_t)qh * S_ + s) * HD_ + d] = (_Float16)(x * cs[d] + xr * sn[d]);
    } else if (idx < (NH_ + NKV_) * HD_) {                    // K (roped, pre-quant)
      const int hh = (idx - NH_ * HD_) >> 7;
      const float x = row[idx];
      const float xr = (d < 64) ? -row[NH_ * HD_ + hh * HD_ + d + 64]
                                :  row[NH_ * HD_ + hh * HD_ + d - 64];
      kf[((size_t)hh * S_ + s) * HD_ + d] = x * cs[d] + xr * sn[d];
    } else {                                                  // V (copy)
      const int hh = (idx - (NH_ + NKV_) * HD_) >> 7;
      vf[((size_t)hh * S_ + s) * HD_ + d] = row[idx];
    }
  }
}

// ---------------------------------------------------------------------------
// K quant/dequant: reference blocks = (h,d) over all t (BLK==kv_len==1024).
// Output k16 [NKV, S(t), HD(d)] f16 (B operand of the scores GEMM).
__global__ void __launch_bounds__(256)
quant_k_kernel(const float* __restrict__ kf, _Float16* __restrict__ k16) {
  const int b = blockIdx.x;                    // h*HD + d
  const int hh = b >> 7, d = b & (HD_ - 1);
  const float* base = kf + (size_t)hh * S_ * HD_ + d;
  __shared__ float smn[256], smx[256];
  float vals[4], mn = 3.4e38f, mx = -3.4e38f;
#pragma unroll
  for (int i = 0; i < 4; ++i) {
    const int t = threadIdx.x + i * 256;
    const float x = base[(size_t)t * HD_];
    vals[i] = x; mn = fminf(mn, x); mx = fmaxf(mx, x);
  }
  smn[threadIdx.x] = mn; smx[threadIdx.x] = mx;
  __syncthreads();
  for (int st = 128; st > 0; st >>= 1) {
    if (threadIdx.x < st) {
      smn[threadIdx.x] = fminf(smn[threadIdx.x], smn[threadIdx.x + st]);
      smx[threadIdx.x] = fmaxf(smx[threadIdx.x], smx[threadIdx.x + st]);
    }
    __syncthreads();
  }
  mn = smn[0]; mx = smx[0];
  const float sc  = (mx - mn) * (1.0f / 255.0f);
  const float inv = 1.0f / (sc + 1e-6f);
#pragma unroll
  for (int i = 0; i < 4; ++i) {
    const int t = threadIdx.x + i * 256;
    const float q = fminf(rintf((vals[i] - mn) * inv), 255.0f);
    k16[((size_t)hh * S_ + t) * HD_ + d] = (_Float16)(q * sc + mn);
  }
}

// V quant/dequant: blocks = 1024 contiguous elements of [NKV, kv, HD].
// Output transposed vT [NKV, HD(d), S(t)] f16 (B operand of the attn GEMM).
__global__ void __launch_bounds__(256)
quant_v_kernel(const float* __restrict__ vfp, _Float16* __restrict__ vT) {
  const size_t b = blockIdx.x;
  const float* base = vfp + b * 1024;
  __shared__ float smn[256], smx[256];
  float vals[4], mn = 3.4e38f, mx = -3.4e38f;
#pragma unroll
  for (int i = 0; i < 4; ++i) {
    const int e = threadIdx.x + i * 256;
    const float x = base[e];
    vals[i] = x; mn = fminf(mn, x); mx = fmaxf(mx, x);
  }
  smn[threadIdx.x] = mn; smx[threadIdx.x] = mx;
  __syncthreads();
  for (int st = 128; st > 0; st >>= 1) {
    if (threadIdx.x < st) {
      smn[threadIdx.x] = fminf(smn[threadIdx.x], smn[threadIdx.x + st]);
      smx[threadIdx.x] = fmaxf(smx[threadIdx.x], smx[threadIdx.x + st]);
    }
    __syncthreads();
  }
  mn = smn[0]; mx = smx[0];
  const float sc  = (mx - mn) * (1.0f / 255.0f);
  const float inv = 1.0f / (sc + 1e-6f);
#pragma unroll
  for (int i = 0; i < 4; ++i) {
    const int e = threadIdx.x + i * 256;
    const size_t g = b * 1024 + e;
    const int hh = (int)(g >> 17);               // / (S_*HD_)
    const int rem = (int)(g & 131071);
    const int t = rem >> 7, d = rem & (HD_ - 1);
    const float q = fminf(rintf((vals[i] - mn) * inv), 255.0f);
    vT[((size_t)hh * HD_ + d) * S_ + t] = (_Float16)(q * sc + mn);
  }
}

// ---------------------------------------------------------------------------
// Causal-masked softmax over kv (reference mask: t <= s allowed, else -128*mf)
__global__ void __launch_bounds__(256)
softmax_kernel(const float* __restrict__ sc, _Float16* __restrict__ pr,
               const int* __restrict__ mfp) {
  const int s = blockIdx.x, hh = blockIdx.y;
  const float* row = sc + ((size_t)hh * S_ + s) * KV_;
  _Float16* prow = pr + ((size_t)hh * S_ + s) * KV_;
  const float neg = -128.0f * (float)(*mfp);
  __shared__ float red[256];
  float vals[4], mx = -3.4e38f;
#pragma unroll
  for (int i = 0; i < 4; ++i) {
    const int t = threadIdx.x + i * 256;
    const float v = row[t] + ((t <= s) ? 0.0f : neg);
    vals[i] = v; mx = fmaxf(mx, v);
  }
  red[threadIdx.x] = mx;
  __syncthreads();
  for (int st = 128; st > 0; st >>= 1) {
    if (threadIdx.x < st) red[threadIdx.x] = fmaxf(red[threadIdx.x], red[threadIdx.x + st]);
    __syncthreads();
  }
  mx = red[0];
  __syncthreads();
  float sum = 0.f;
#pragma unroll
  for (int i = 0; i < 4; ++i) { vals[i] = expf(vals[i] - mx); sum += vals[i]; }
  red[threadIdx.x] = sum;
  __syncthreads();
  for (int st = 128; st > 0; st >>= 1) {
    if (threadIdx.x < st) red[threadIdx.x] += red[threadIdx.x + st];
    __syncthreads();
  }
  const float inv = 1.0f / red[0];
#pragma unroll
  for (int i = 0; i < 4; ++i) {
    const int t = threadIdx.x + i * 256;
    prow[t] = (_Float16)(vals[i] * inv);
  }
}

// silu(g) * u -> f16
__global__ void __launch_bounds__(256)
silu_mul_kernel(const float* __restrict__ g, const float* __restrict__ u,
                _Float16* __restrict__ x) {
  const size_t i = (size_t)blockIdx.x * 256 + threadIdx.x;
  const float gv = g[i];
  const float s = gv / (1.0f + expf(-gv));
  x[i] = (_Float16)(s * u[i]);
}

// LM head: logits[n] = hn[last] . w_lm[n]  (fp32 dot; bandwidth bound)
__global__ void __launch_bounds__(256)
lmhead_kernel(const float* __restrict__ hn, const float* __restrict__ wlm,
              float* __restrict__ out) {
  const int wave = threadIdx.x >> 5, lane = threadIdx.x & 31;
  const int n = blockIdx.x * 8 + wave;
  const float* hrow = hn + (size_t)(S_ - 1) * D_;
  const float* w = wlm + (size_t)n * D_;
  float acc = 0.f;
  for (int d = lane * 4; d < D_; d += 128) {
    const float4 a = *(const float4*)(hrow + d);
    const float4 b = *(const float4*)(w + d);
    acc += a.x * b.x + a.y * b.y + a.z * b.z + a.w * b.w;
  }
#pragma unroll
  for (int o = 16; o > 0; o >>= 1) acc += __shfl_xor(acc, o, 32);
  if (lane == 0) out[n] = acc;
}

// ---------------------------------------------------------------------------
extern "C" void kernel_launch(void* const* d_in, const int* in_sizes, int n_in,
                              void* d_out, int out_size, void* d_ws, size_t ws_size,
                              hipStream_t stream) {
  (void)in_sizes; (void)n_in; (void)out_size; (void)ws_size;
  const float* x_in   = (const float*)d_in[0];
  const float* w_qkv  = (const float*)d_in[1];
  const float* w_o    = (const float*)d_in[2];
  const float* w_gate = (const float*)d_in[3];
  const float* w_up   = (const float*)d_in[4];
  const float* w_down = (const float*)d_in[5];
  const float* w_lm   = (const float*)d_in[6];
  const float* cosT   = (const float*)d_in[7];
  const float* sinT   = (const float*)d_in[8];
  const int*   histp  = (const int*)d_in[9];
  const int*   mfp    = (const int*)d_in[11];
  float* logits = (float*)d_out;

  // Workspace layout (~151 MB, 256B aligned chunks)
  char* W = (char*)d_ws;
  size_t off = 0;
  auto take = [&](size_t bytes) -> char* {
    char* c = W + off;
    off += (bytes + 255) & ~(size_t)255;
    return c;
  };
  float*    h       = (float*)   take((size_t)S_ * D_ * 4);
  _Float16* hn16    = (_Float16*)take((size_t)S_ * D_ * 2);
  float*    qkv     = (float*)   take((size_t)S_ * QKVW_ * 4);
  _Float16* q16     = (_Float16*)take((size_t)NH_ * S_ * HD_ * 2);
  float*    kf      = (float*)   take((size_t)NKV_ * S_ * HD_ * 4);
  float*    vf      = (float*)   take((size_t)NKV_ * S_ * HD_ * 4);
  _Float16* k16     = (_Float16*)take((size_t)NKV_ * S_ * HD_ * 2);
  _Float16* vT16    = (_Float16*)take((size_t)NKV_ * S_ * HD_ * 2);
  _Float16* attn16  = (_Float16*)take((size_t)S_ * D_ * 2);
  float*    scores  = (float*)   take((size_t)NH_ * S_ * KV_ * 4);   // 64 MB
  _Float16* probs16 = (_Float16*)take((size_t)NH_ * S_ * KV_ * 2);   // 32 MB
  // Aliases (lifetimes disjoint): gate/up reuse scores; silu out reuses probs;
  // final fp32 hn reuses qkv.
  float*    g_buf  = scores;
  float*    u_buf  = scores + (size_t)S_ * FF_;
  _Float16* x16    = probs16;
  float*    hnf32  = qkv;

  for (int i = 0; i < L_; ++i) {
    const float* hin = (i == 0) ? x_in : h;

    // hn = rms(h) -> f16
    rmsnorm_kernel<<<S_, 256, 0, stream>>>(hin, hn16, nullptr);

    // qkv = hn @ w_qkv^T   [1024 x 4096], K=2048
    wgemm<true, false, false><<<dim3(QKVW_ / 128, S_ / 128, 1), 256, 0, stream>>>(
        hn16, w_qkv + (size_t)i * QKVW_ * D_, qkv, nullptr,
        D_, D_, D_, QKVW_, 0, 0, 0, 1);

    // RoPE + split
    rope_split_kernel<<<S_, 256, 0, stream>>>(qkv, cosT, sinT, histp, q16, kf, vf);

    // int8 round-trip of K and V
    quant_k_kernel<<<NKV_ * HD_, 256, 0, stream>>>(kf, k16);
    quant_v_kernel<<<(NKV_ * S_ * HD_) / 1024, 256, 0, stream>>>(vf, vT16);

    // scores[h] = q[h] @ k[h/2]^T   per head: [1024 x 1024], K=128
    wgemm<false, false, false><<<dim3(KV_ / 128, S_ / 128, NH_), 256, 0, stream>>>(
        q16, k16, scores, nullptr,
        HD_, HD_, HD_, KV_,
        (long)S_ * HD_, (long)S_ * HD_, (long)S_ * KV_, NH_ / NKV_);

    // masked softmax -> probs f16
    softmax_kernel<<<dim3(S_, NH_), 256, 0, stream>>>(scores, probs16, mfp);

    // attn[h] = probs[h] @ v[h/2]   per head: [1024 x 128], K=1024
    wgemm<false, false, true><<<dim3(HD_ / 128, S_ / 128, NH_), 256, 0, stream>>>(
        probs16, vT16, attn16, nullptr,
        KV_, KV_, KV_, D_,
        (long)S_ * KV_, (long)HD_ * S_, (long)HD_, NH_ / NKV_);

    // h = hin + attn @ w_o^T
    wgemm<true, true, false><<<dim3(D_ / 128, S_ / 128, 1), 256, 0, stream>>>(
        attn16, w_o + (size_t)i * D_ * D_, h, hin,
        D_, D_, D_, D_, 0, 0, 0, 1);

    // hn2 = rms(h)
    rmsnorm_kernel<<<S_, 256, 0, stream>>>(h, hn16, nullptr);

    // gate / up projections [1024 x 6144], K=2048
    wgemm<true, false, false><<<dim3(FF_ / 128, S_ / 128, 1), 256, 0, stream>>>(
        hn16, w_gate + (size_t)i * FF_ * D_, g_buf, nullptr,
        D_, D_, D_, FF_, 0, 0, 0, 1);
    wgemm<true, false, false><<<dim3(FF_ / 128, S_ / 128, 1), 256, 0, stream>>>(
        hn16, w_up + (size_t)i * FF_ * D_, u_buf, nullptr,
        D_, D_, D_, FF_, 0, 0, 0, 1);

    // x = silu(g) * u  -> f16
    silu_mul_kernel<<<(S_ * FF_) / 256, 256, 0, stream>>>(g_buf, u_buf, x16);

    // h = h + x @ w_down^T   [1024 x 2048], K=6144
    wgemm<true, true, false><<<dim3(D_ / 128, S_ / 128, 1), 256, 0, stream>>>(
        x16, w_down + (size_t)i * D_ * FF_, h, h,
        FF_, FF_, FF_, D_, 0, 0, 0, 1);
  }

  // final norm (fp32) + LM head on last token
  rmsnorm_kernel<<<S_, 256, 0, stream>>>(h, nullptr, hnf32);
  lmhead_kernel<<<V_ / 8, 256, 0, stream>>>(hnf32, w_lm, logits);
}
